// DeepFusion_2207613190691
// MI455X (gfx1250) — matrix operand order repaired and checked by reading, not compiled, not run on hardware
//
#include <hip/hip_runtime.h>

typedef __bf16  v16bf __attribute__((ext_vector_type(16)));
typedef __bf16  v8bf  __attribute__((ext_vector_type(8)));
typedef float   v8f   __attribute__((ext_vector_type(8)));

__device__ __forceinline__ float selu_f(float x) {
  const float a = 1.6732632423543772f, s = 1.0507009873554805f;
  return x > 0.f ? s * x : s * a * (__expf(x) - 1.f);
}

// ---------------------------------------------------------------------------
// Pack conv weights (cout,cin,k,k) f32 -> bf16 in per-lane WMMA B-fragment
// order: bpack[((nTile*nKTiles + t)*32 + lane)*16 + c] holds
//   K = t*32 + (lane>=16 ? 8 : 0) + (c<8 ? c : c+8) ,  N = nTile*16 + (lane&15)
// with K decomposed as ((ky*kk + kx)*cin + ci)  (matches NHWC activations).
// Padded K / N entries are zero (lets the conv loop run guard-free).
// ---------------------------------------------------------------------------
__global__ void pack_weights(const float* __restrict__ w, __bf16* __restrict__ bpack,
                             int cin, int cout, int kk, int Kreal,
                             int nKTiles, int nNTiles) {
  int id = blockIdx.x * blockDim.x + threadIdx.x;
  int total = nNTiles * nKTiles * 512;
  if (id >= total) return;
  int c    = id & 15;
  int lane = (id >> 4) & 31;
  int blk  = id >> 9;
  int t    = blk % nKTiles;
  int nT   = blk / nKTiles;
  int n = nT * 16 + (lane & 15);
  int k = t * 32 + ((lane & 16) ? 8 : 0) + (c < 8 ? c : c + 8);
  float val = 0.f;
  if (k < Kreal && n < cout) {
    int ci = k % cin;
    int q  = k / cin;
    int ky = q / kk;
    int kx = q - ky * kk;
    val = w[(((size_t)n * cin + ci) * kk + ky) * kk + kx];
  }
  bpack[id] = (__bf16)val;
}

// ---------------------------------------------------------------------------
// Implicit-GEMM conv via v_wmma_f32_16x16x32_bf16.
// One wave = one 16(M pixels) x (NT*16)(cout) strip; K looped in steps of 32.
// The K->input-offset map is precomputed into LDS once per block, so the hot
// loop is pure loads + WMMA (no div/mod, no exec masking).
//   MODE 0: input NCHW f32 (layer 1), scalar gather
//   MODE 1: input NHWC bf16, scalar gather (cin not a multiple of 8)
//   MODE 2: input NHWC bf16, cin % 8 == 0 -> two 16B vector loads per K-step
// A fragment per ISA layout: lane<16 -> K {0..7,16..23}; lane>=16 -> K {8..15,24..31}.
// C layout: lane<16 rows r, lane>=16 rows r+8, N = lane&15.
// ---------------------------------------------------------------------------
template <int NT, int MODE>
__global__ void __launch_bounds__(256)
conv_wmma_t(const void* __restrict__ inPtr,
            int inH, int inW, int cin, int cout, int kk, int stride,
            int outW, int Mtot, int Kreal, int nKTiles,
            const __bf16* __restrict__ bpack,
            const float* __restrict__ bias,
            __bf16* __restrict__ outBf,        // NHWC bf16 (or null)
            float* __restrict__ outF32,        // NHWC f32  (or null)
            int useRelu) {
  __shared__ int sOff[896];

  // Build the K -> input-element-offset table (once per block).
  const int Kpad = nKTiles * 32;
  const int nEnt = (MODE == 2) ? (Kpad >> 3) : Kpad;
  for (int e = threadIdx.x; e < nEnt; e += 256) {
    const int k = (MODE == 2) ? (e << 3) : e;
    int off = 0;
    if (k < Kreal) {
      const int ci = k % cin;
      const int q  = k / cin;
      const int ky = q / kk;
      const int kx = q - ky * kk;
      off = (MODE == 0) ? (ci * inH * inW + ky * inW + kx)
                        : ((ky * inW + kx) * cin + ci);
    }
    sOff[e] = off;
  }
  __syncthreads();

  const int wave    = threadIdx.x >> 5;
  const int lane    = threadIdx.x & 31;
  const int nMTiles = (Mtot + 15) >> 4;
  const int mTile   = blockIdx.x * 8 + wave;
  if (mTile >= nMTiles) return;           // wave-uniform: EXEC stays all-1s
  const int b = blockIdx.y;

  const int nLoc  = lane & 15;
  const int kASel = (lane & 16) ? 8 : 0;
  int pixA = mTile * 16 + nLoc;
  pixA = pixA < Mtot ? pixA : Mtot - 1;   // clamp: invalid rows never stored
  const int oy = pixA / outW;
  const int ox = pixA - oy * outW;
  const int pixBase = (MODE == 0) ? ((oy * stride) * inW + (ox * stride))
                                  : ((oy * stride) * inW + (ox * stride)) * cin;

  const float*  inF = (const float*)inPtr  + (size_t)b * cin * inH * inW + pixBase;
  const __bf16* inB = (const __bf16*)inPtr + (size_t)b * cin * inH * inW + pixBase;

  const v8f vzero = {0.f, 0.f, 0.f, 0.f, 0.f, 0.f, 0.f, 0.f};
  v8f acc[NT];
#pragma unroll
  for (int nt = 0; nt < NT; ++nt) acc[nt] = vzero;

  for (int t = 0; t < nKTiles; ++t) {
    v16bf a;
    if (MODE == 2) {
      const int e0 = (t * 32 + kASel) >> 3;  // two contiguous 8-elem K runs
      const int o0 = sOff[e0];
      const int o1 = sOff[e0 + 2];
      const v8bf lo = *(const v8bf*)(inB + o0);
      const v8bf hi = *(const v8bf*)(inB + o1);
      a = __builtin_shufflevector(lo, hi, 0, 1, 2, 3, 4, 5, 6, 7,
                                  8, 9, 10, 11, 12, 13, 14, 15);
    } else {
#pragma unroll
      for (int e = 0; e < 16; ++e) {
        const int kl  = (e < 8 ? e : e + 8) + kASel;
        const int off = sOff[t * 32 + kl];
        const float val = (MODE == 0) ? inF[off] : (float)inB[off];
        a[e] = (__bf16)val;
      }
    }
    const __bf16* bpt = bpack + ((size_t)t << 9) + (lane << 4);
    if (t + 1 < nKTiles)
      __builtin_prefetch(bpt + 512, 0, 1);
#pragma unroll
    for (int nt = 0; nt < NT; ++nt) {
      const v16bf bf = *(const v16bf*)(bpt + (((size_t)nt * nKTiles) << 9));
      acc[nt] = __builtin_amdgcn_wmma_f32_16x16x32_bf16(false, a, false, bf,
                                                        (short)0, acc[nt],
                                                        false, false);
    }
  }

  const size_t obase = (size_t)b * Mtot * cout;
#pragma unroll
  for (int nt = 0; nt < NT; ++nt) {
    const int n = nt * 16 + nLoc;
    const float bv = (n < cout) ? bias[n] : 0.f;
#pragma unroll
    for (int r = 0; r < 8; ++r) {
      const int m   = r + ((lane & 16) ? 8 : 0);
      const int pix = mTile * 16 + m;
      if (pix < Mtot && n < cout) {
        float v = acc[nt][r] + bv;
        v = useRelu ? fmaxf(v, 0.f) : selu_f(v);
        if (outF32) outF32[obase + (size_t)pix * cout + n] = v;
        else        outBf [obase + (size_t)pix * cout + n] = (__bf16)v;
      }
    }
  }
}

// ---------------------------------------------------------------------------
// Adaptive avg pool 117x117 -> 16x16, NHWC f32 input, output [b][c][i][j].
// ---------------------------------------------------------------------------
__global__ void pool_adaptive(const float* __restrict__ act, float* __restrict__ pooled) {
  int id = blockIdx.x * blockDim.x + threadIdx.x;
  if (id >= 2 * 24 * 16 * 16) return;
  int j = id & 15, i = (id >> 4) & 15, c = (id >> 8) % 24, b = id / (24 * 256);
  int sh = (i * 117) >> 4, eh = ((i + 1) * 117 + 15) >> 4;
  int sw = (j * 117) >> 4, ew = ((j + 1) * 117 + 15) >> 4;
  const float* base = act + (size_t)b * 117 * 117 * 24;
  float s = 0.f;
  for (int y = sh; y < eh; ++y)
    for (int x = sw; x < ew; ++x)
      s += base[((size_t)y * 117 + x) * 24 + c];
  pooled[id] = s / (float)((eh - sh) * (ew - sw));
}

// ---------------------------------------------------------------------------
// grid / max(sum_e |grid|, 1e-12): pooled[b][c=e*8+ib][i][j] -> ng[b][e][ib][i][j]
// ---------------------------------------------------------------------------
__global__ void grid_normalize(const float* __restrict__ pooled, float* __restrict__ ng) {
  int id = blockIdx.x * blockDim.x + threadIdx.x;
  if (id >= 2 * 8 * 16 * 16) return;
  int j = id & 15, i = (id >> 4) & 15, ib = (id >> 8) & 7, b = id >> 11;
  float v[3]; float d = 0.f;
#pragma unroll
  for (int e = 0; e < 3; ++e) {
    v[e] = pooled[((((size_t)b * 24 + (e * 8 + ib)) * 16 + i) << 4) + j];
    d += fabsf(v[e]);
  }
  d = fmaxf(d, 1e-12f);
#pragma unroll
  for (int e = 0; e < 3; ++e)
    ng[((((size_t)(b * 3 + e) * 8 + ib) * 16 + i) << 4) + j] = v[e] / d;
}

// ---------------------------------------------------------------------------
// Fused: guide net (1x1 conv -> SELU -> 1x1 conv -> tanh), trilinear grid
// slice, and per-pixel recombination  out[b,r] = sum_e inp[b,e*3+r]*coeff[e].
// Guide weights staged in LDS.
// ---------------------------------------------------------------------------
__global__ void __launch_bounds__(256)
guide_slice_apply(const float* __restrict__ inp,
                  const float* __restrict__ w0, const float* __restrict__ b0,
                  const float* __restrict__ w1, const float* __restrict__ b1,
                  const float* __restrict__ ng, float* __restrict__ out) {
  __shared__ float sW0[144], sB0[16], sW1[16], sB1[1];
  const int tid = threadIdx.x;
  if (tid < 144) sW0[tid] = w0[tid];
  if (tid < 16) { sB0[tid] = b0[tid]; sW1[tid] = w1[tid]; }
  if (tid == 0) sB1[0] = b1[0];
  __syncthreads();

  const size_t HW = 1024u * 1024u;
  size_t id = (size_t)blockIdx.x * blockDim.x + tid;
  if (id >= 2 * HW) return;
  const int b   = (int)(id / HW);
  const int rem = (int)(id % HW);
  const int y   = rem >> 10;
  const int x   = rem & 1023;

  const float* ip = inp + (size_t)b * 9 * HW + rem;
  float v[9];
#pragma unroll
  for (int c = 0; c < 9; ++c) v[c] = ip[(size_t)c * HW];

  // guide network
  float g = sB1[0];
#pragma unroll
  for (int h = 0; h < 16; ++h) {
    float hh = sB0[h];
#pragma unroll
    for (int c = 0; c < 9; ++c) hh = fmaf(sW0[h * 9 + c], v[c], hh);
    g = fmaf(sW1[h], selu_f(hh), g);
  }
  const float z = tanhf(g);

  // trilinear slice of the bilateral grid
  const float ix = 16.f * (float)x / 1023.f - 0.5f;
  const float iy = 16.f * (float)y / 1023.f - 0.5f;
  const float iz = ((z + 1.f) * 8.f - 1.f) * 0.5f;
  const float x0 = floorf(ix), y0 = floorf(iy), z0 = floorf(iz);
  float coeff[3] = {0.f, 0.f, 0.f};
#pragma unroll
  for (int dz = 0; dz < 2; ++dz)
#pragma unroll
    for (int dy = 0; dy < 2; ++dy)
#pragma unroll
      for (int dx = 0; dx < 2; ++dx) {
        const float xc = x0 + dx, yc = y0 + dy, zc = z0 + dz;
        float wt = (1.f - fabsf(ix - xc)) * (1.f - fabsf(iy - yc)) * (1.f - fabsf(iz - zc));
        const bool inb = (xc >= 0.f) && (xc < 16.f) && (yc >= 0.f) && (yc < 16.f) &&
                         (zc >= 0.f) && (zc < 8.f);
        wt = inb ? wt : 0.f;
        const int xi = min(max((int)xc, 0), 15);
        const int yi = min(max((int)yc, 0), 15);
        const int zi = min(max((int)zc, 0), 7);
#pragma unroll
        for (int e = 0; e < 3; ++e)
          coeff[e] = fmaf(ng[((((size_t)(b * 3 + e) * 8 + zi) * 16 + yi) << 4) + xi], wt, coeff[e]);
      }

#pragma unroll
  for (int r = 0; r < 3; ++r) {
    float o = 0.f;
#pragma unroll
    for (int e = 0; e < 3; ++e) o = fmaf(v[e * 3 + r], coeff[e], o);
    out[(size_t)(b * 3 + r) * HW + rem] = o;
  }
}

// ---------------------------------------------------------------------------
extern "C" void kernel_launch(void* const* d_in, const int* in_sizes, int n_in,
                              void* d_out, int out_size, void* d_ws, size_t ws_size,
                              hipStream_t stream) {
  (void)in_sizes; (void)n_in; (void)out_size; (void)ws_size;
  const float* x = (const float*)d_in[0];
  // d_in order: exposure_stack, grid_ws[0..7], grid_bs[0..7], guide_ws[0..1], guide_bs[0..1]

  struct LS { int cin, cout, kk, st, inH, inW, outH, outW; };
  static const LS L[8] = {
    { 9, 12, 7, 2, 1024, 1024, 509, 509},
    {12, 24, 3, 2,  509,  509, 254, 254},
    {24, 24, 3, 1,  254,  254, 252, 252},
    {24, 48, 3, 2,  252,  252, 125, 125},
    {48, 48, 3, 1,  125,  125, 123, 123},
    {48, 96, 3, 1,  123,  123, 121, 121},
    {96, 96, 3, 1,  121,  121, 119, 119},
    {96, 24, 3, 1,  119,  119, 117, 117}};

  char* wsb = (char*)d_ws;
  size_t off = 0;
  auto take = [&](size_t bytes) -> void* {
    void* p = wsb + off;
    off = (off + bytes + 255) & ~(size_t)255;
    return p;
  };

  __bf16* actB[7];
  float* act8 = nullptr;
  for (int i = 0; i < 8; ++i) {
    size_t elems = (size_t)2 * L[i].outH * L[i].outW * L[i].cout;
    if (i < 7) actB[i] = (__bf16*)take(elems * 2);
    else       act8    = (float*)take(elems * 4);
  }
  __bf16* bp[8];
  int nK[8], nN[8], Kr[8];
  for (int i = 0; i < 8; ++i) {
    Kr[i] = L[i].cin * L[i].kk * L[i].kk;
    nK[i] = (Kr[i] + 31) / 32;
    nN[i] = (L[i].cout + 15) / 16;
    bp[i] = (__bf16*)take((size_t)nN[i] * nK[i] * 512 * 2);
  }
  float* pooled = (float*)take((size_t)2 * 24 * 256 * 4);
  float* ng     = (float*)take((size_t)2 * 3 * 8 * 256 * 4);

  for (int i = 0; i < 8; ++i) {
    int total = nN[i] * nK[i] * 512;
    pack_weights<<<(total + 255) / 256, 256, 0, stream>>>(
        (const float*)d_in[1 + i], bp[i],
        L[i].cin, L[i].cout, L[i].kk, Kr[i], nK[i], nN[i]);

    int Mtot = L[i].outH * L[i].outW;
    int nM   = (Mtot + 15) / 16;
    dim3 grid((nM + 7) / 8, 2);
    const void* inPtr = (i == 0) ? (const void*)x : (const void*)actB[i - 1];
    __bf16* ob  = (i < 7) ? actB[i] : nullptr;
    float*  of  = (i == 7) ? act8 : nullptr;
    int     rl  = (i == 7) ? 1 : 0;
    const float* bb = (const float*)d_in[9 + i];

#define CONV_ARGS inPtr, L[i].inH, L[i].inW, L[i].cin, L[i].cout, L[i].kk, \
                  L[i].st, L[i].outW, Mtot, Kr[i], nK[i], bp[i], bb, ob, of, rl
    switch (i) {
      case 0: conv_wmma_t<1, 0><<<grid, 256, 0, stream>>>(CONV_ARGS); break;
      case 1: conv_wmma_t<2, 1><<<grid, 256, 0, stream>>>(CONV_ARGS); break;
      case 2: conv_wmma_t<2, 2><<<grid, 256, 0, stream>>>(CONV_ARGS); break;
      case 3: conv_wmma_t<3, 2><<<grid, 256, 0, stream>>>(CONV_ARGS); break;
      case 4: conv_wmma_t<3, 2><<<grid, 256, 0, stream>>>(CONV_ARGS); break;
      case 5: conv_wmma_t<6, 2><<<grid, 256, 0, stream>>>(CONV_ARGS); break;
      case 6: conv_wmma_t<6, 2><<<grid, 256, 0, stream>>>(CONV_ARGS); break;
      case 7: conv_wmma_t<2, 2><<<grid, 256, 0, stream>>>(CONV_ARGS); break;
    }
#undef CONV_ARGS
  }

  pool_adaptive<<<48, 256, 0, stream>>>(act8, pooled);
  grid_normalize<<<16, 256, 0, stream>>>(pooled, ng);
  guide_slice_apply<<<8192, 256, 0, stream>>>(
      x,
      (const float*)d_in[17], (const float*)d_in[19],
      (const float*)d_in[18], (const float*)d_in[20],
      ng, (float*)d_out);
}
